// hsdn_74509092651634
// MI455X (gfx1250) — compile-verified, matrix operand order checked by reading, not run on hardware
//
#include <hip/hip_runtime.h>
#include <hip/hip_bf16.h>

#define N_NODES 20000
#define M_EDGES 20000
#define E_INC   400000
#define F_DIM   256
#define HID     256
#define HEADS   4
#define FS      64
#define OUT_DIM 128

typedef __attribute__((ext_vector_type(16))) __bf16 v16bf;
typedef __attribute__((ext_vector_type(8)))  float  v8f;

__device__ __forceinline__ float wave_reduce_sum(float v) {
#pragma unroll
    for (int m = 16; m > 0; m >>= 1) v += __shfl_xor(v, m, 32);
    return v;
}
__device__ __forceinline__ float wave_reduce_max(float v) {
#pragma unroll
    for (int m = 16; m > 0; m >>= 1) v = fmaxf(v, __shfl_xor(v, m, 32));
    return v;
}

// ---------------------------------------------------------------------------
// One-time weight pre-pass: convert f32 B[K x Ncols] to bf16, swizzled into
// per-(ktile, ntile, lane) WMMA B-fragments so the GEMM reads each lane's
// fragment as a single 32B vector. Fragment layout per CDNA5 ISA:
//   lane: n = lane&15, khi = (lane>>4)*8
//   elem 2v+j : K = (v>>2)*16 + khi + (v&3)*2 + j
// ---------------------------------------------------------------------------
__global__ void k_pack_b(const float* __restrict__ B, v16bf* __restrict__ Bswz,
                         int K, int Ncols) {
    const int lane  = threadIdx.x & 31;
    const int wave  = threadIdx.x >> 5;
    const int ntn   = Ncols >> 4;
    const int tiles = (K >> 5) * ntn;
    const int tile  = blockIdx.x * (blockDim.x >> 5) + wave;
    if (tile >= tiles) return;
    const int kt  = (tile / ntn) << 5;
    const int tn  = (tile % ntn) << 4;
    const int lm  = lane & 15;
    const int khi = (lane >> 4) * 8;
    v16bf b;
#pragma unroll
    for (int v = 0; v < 8; ++v) {
        const int kb = ((v >> 2) << 4) + khi + ((v & 3) << 1);
        b[2 * v]     = (__bf16)B[(kt + kb) * Ncols + tn + lm];
        b[2 * v + 1] = (__bf16)B[(kt + kb + 1) * Ncols + tn + lm];
    }
    Bswz[(size_t)tile * 32 + lane] = b;
}

// Per-lane A fragment = two contiguous 8-float chunks starting at k0, k0+16.
__device__ __forceinline__ v16bf load_a_frag(const float* __restrict__ Arow, int k0) {
    const float4 a0 = *(const float4*)(Arow + k0);
    const float4 a1 = *(const float4*)(Arow + k0 + 4);
    const float4 a2 = *(const float4*)(Arow + k0 + 16);
    const float4 a3 = *(const float4*)(Arow + k0 + 20);
    v16bf a;
    a[0]  = (__bf16)a0.x; a[1]  = (__bf16)a0.y; a[2]  = (__bf16)a0.z; a[3]  = (__bf16)a0.w;
    a[4]  = (__bf16)a1.x; a[5]  = (__bf16)a1.y; a[6]  = (__bf16)a1.z; a[7]  = (__bf16)a1.w;
    a[8]  = (__bf16)a2.x; a[9]  = (__bf16)a2.y; a[10] = (__bf16)a2.z; a[11] = (__bf16)a2.w;
    a[12] = (__bf16)a3.x; a[13] = (__bf16)a3.y; a[14] = (__bf16)a3.z; a[15] = (__bf16)a3.w;
    return a;
}

// ---------------------------------------------------------------------------
// WMMA bf16 GEMM: C[Mrows x Ncols] = A[Mrows x K] * B(pre-swizzled bf16).
// Mrows%16==0, K%64==0, Ncols%16==0. One 16x16 tile per wave, K unrolled x2.
// ---------------------------------------------------------------------------
__global__ void k_gemm_bf16(const float* __restrict__ A, const v16bf* __restrict__ Bswz,
                            float* __restrict__ C, int Mrows, int K, int Ncols) {
    const int lane  = threadIdx.x & 31;
    const int wave  = threadIdx.x >> 5;
    const int ntn   = Ncols >> 4;
    const int tiles = (Mrows >> 4) * ntn;
    const int tile  = blockIdx.x * (blockDim.x >> 5) + wave;
    if (tile >= tiles) return;
    const int tm  = (tile / ntn) << 4;
    const int tnT = tile % ntn;
    const int lm  = lane & 15;
    const int khi = (lane >> 4) * 8;

    const float* Arow = A + (size_t)(tm + lm) * K;
    v8f c = {};
    for (int kk = 0; kk < K; kk += 64) {
        const v16bf b0 = Bswz[(size_t)(((kk >> 5) + 0) * ntn + tnT) * 32 + lane];
        const v16bf a0 = load_a_frag(Arow, kk + khi);
        c = __builtin_amdgcn_wmma_f32_16x16x32_bf16(false, a0, false, b0,
                                                    (short)0, c, false, false);
        const v16bf b1 = Bswz[(size_t)(((kk >> 5) + 1) * ntn + tnT) * 32 + lane];
        const v16bf a1 = load_a_frag(Arow, kk + 32 + khi);
        c = __builtin_amdgcn_wmma_f32_16x16x32_bf16(false, a1, false, b1,
                                                    (short)0, c, false, false);
    }
    const int row0 = tm + ((lane >> 4) << 3);
#pragma unroll
    for (int r = 0; r < 8; ++r)
        C[(size_t)(row0 + r) * Ncols + (tnT << 4) + lm] = c[r];
}

// Per-(node,head) softmax over FS=64 contiguous values. One wave per row.
__global__ void k_softmax(float* __restrict__ h, int rows) {
    const int lane = threadIdx.x & 31;
    const int row  = blockIdx.x * (blockDim.x >> 5) + (threadIdx.x >> 5);
    if (row >= rows) return;
    float* p = h + (size_t)row * FS;
    float v0 = p[lane], v1 = p[lane + 32];
    float mx = wave_reduce_max(fmaxf(v0, v1));
    float e0 = __expf(v0 - mx), e1 = __expf(v1 - mx);
    float inv = 1.0f / wave_reduce_sum(e0 + e1);
    p[lane] = e0 * inv;
    p[lane + 32] = e1 * inv;
}

// Incidence counts per hyperedge and per node.
__global__ void k_counts(const int* __restrict__ src, const int* __restrict__ he,
                         float* __restrict__ cnt_node, float* __restrict__ cnt_he, int E) {
    int e = blockIdx.x * blockDim.x + threadIdx.x;
    if (e >= E) return;
    atomicAdd(&cnt_he[he[e]], 1.0f);
    atomicAdd(&cnt_node[src[e]], 1.0f);
}

// out[sidx[e]][:] += vals[gidx[e]][:], C=256, float4 per thread (64 threads),
// grid-stride with data-dependent prefetch of the next gather row.
__global__ void k_scatter_add_rows(const float* __restrict__ vals,
                                   const int* __restrict__ gidx,
                                   const int* __restrict__ sidx,
                                   float* __restrict__ out, int E) {
    const int c4 = threadIdx.x;  // 0..63 -> columns 4*c4..4*c4+3
    for (int e = blockIdx.x; e < E; e += gridDim.x) {
        const int en = e + gridDim.x;
        if (en < E)
            __builtin_prefetch(&vals[(size_t)gidx[en] * 256 + c4 * 4], 0, 1);
        const int g = gidx[e], s = sidx[e];
        const float4 v = *(const float4*)&vals[(size_t)g * 256 + c4 * 4];
        float* o = &out[(size_t)s * 256 + c4 * 4];
        atomicAdd(o + 0, v.x);
        atomicAdd(o + 1, v.y);
        atomicAdd(o + 2, v.z);
        atomicAdd(o + 3, v.w);
    }
}

// data[r][:] /= max(cnt[r],1)
__global__ void k_div_rows(float* __restrict__ data, const float* __restrict__ cnt,
                           int C) {
    const int r = blockIdx.x;
    const float inv = 1.0f / fmaxf(cnt[r], 1.0f);
    for (int c = threadIdx.x; c < C; c += blockDim.x)
        data[(size_t)r * C + c] *= inv;
}

// Per incidence: 4 head-dots of length 64, scatter-min (ordered-uint) per hyperedge.
__global__ void k_att(const float* __restrict__ temp_edge, const float* __restrict__ h,
                      const int* __restrict__ src, const int* __restrict__ he,
                      unsigned* __restrict__ att_u, int E) {
    __shared__ float red[256];
    int e = blockIdx.x;
    if (e >= E) return;
    int t = threadIdx.x;
    int m = he[e], n = src[e];
    red[t] = temp_edge[(size_t)m * HID + t] * h[(size_t)n * HID + t];
    __syncthreads();
    for (int s = 32; s > 0; s >>= 1) {
        if ((t & 63) < s) red[t] += red[t + s];
        __syncthreads();
    }
    if ((t & 63) == 0) {
        int head = t >> 6;
        unsigned u = __float_as_uint(red[t]);
        u = (u & 0x80000000u) ? ~u : (u | 0x80000000u);  // order-preserving encode
        atomicMin(&att_u[(size_t)m * HEADS + head], u);
    }
}

__global__ void k_att_decode(const unsigned* __restrict__ att_u,
                             float* __restrict__ att, int total) {
    int i = blockIdx.x * blockDim.x + threadIdx.x;
    if (i >= total) return;
    unsigned u = att_u[i];
    float v;
    if (u == 0xFFFFFFFFu) v = 0.0f;  // empty hyperedge: inf -> 0
    else v = (u & 0x80000000u) ? __uint_as_float(u ^ 0x80000000u) : __uint_as_float(~u);
    att[i] = v;
}

// Per hyperedge: 4x4 Gram over [4,64], loss = mean(-diag + logsumexp(row)).
__global__ void k_factor_loss(const float* __restrict__ temp_edge,
                              float* __restrict__ loss, int M) {
    const int lane = threadIdx.x & 31;
    const int m = blockIdx.x * (blockDim.x >> 5) + (threadIdx.x >> 5);
    if (m >= M) return;
    const float* p = temp_edge + (size_t)m * HID;
    float a0[HEADS], a1[HEADS];
#pragma unroll
    for (int i = 0; i < HEADS; ++i) {
        const float2 v = *(const float2*)&p[i * FS + 2 * lane];
        a0[i] = v.x;
        a1[i] = v.y;
    }
    float gram[HEADS][HEADS];
#pragma unroll
    for (int i = 0; i < HEADS; ++i)
#pragma unroll
        for (int j = 0; j < HEADS; ++j)
            gram[i][j] = wave_reduce_sum(a0[i] * a0[j] + a1[i] * a1[j]);
    if (lane == 0) {
        float part = 0.0f;
#pragma unroll
        for (int i = 0; i < HEADS; ++i) {
            float mx = gram[i][0];
#pragma unroll
            for (int j = 1; j < HEADS; ++j) mx = fmaxf(mx, gram[i][j]);
            float s = 0.0f;
#pragma unroll
            for (int j = 0; j < HEADS; ++j) s += __expf(gram[i][j] - mx);
            part += -gram[i][i] + (mx + __logf(s));
        }
        atomicAdd(loss, part * (1.0f / (float)(M * HEADS)));
    }
}

// x_out[src[e]][head*256+c] += att[he[e]][head] * xe[he[e]][c]; float4/thread.
__global__ void k_prop(const float* __restrict__ xe, const float* __restrict__ att,
                       const int* __restrict__ src, const int* __restrict__ he,
                       float* __restrict__ x_out, int E) {
    const int c4 = threadIdx.x;  // 0..63
    for (int e = blockIdx.x; e < E; e += gridDim.x) {
        const int en = e + gridDim.x;
        if (en < E)
            __builtin_prefetch(&xe[(size_t)he[en] * F_DIM + c4 * 4], 0, 1);
        const int m = he[e], n = src[e];
        const float4 v = *(const float4*)&xe[(size_t)m * F_DIM + c4 * 4];
#pragma unroll
        for (int head = 0; head < HEADS; ++head) {
            const float s = att[(size_t)m * HEADS + head];
            float* o = &x_out[(size_t)n * (HEADS * F_DIM) + head * F_DIM + c4 * 4];
            atomicAdd(o + 0, s * v.x);
            atomicAdd(o + 1, s * v.y);
            atomicAdd(o + 2, s * v.z);
            atomicAdd(o + 3, s * v.w);
        }
    }
}

extern "C" void kernel_launch(void* const* d_in, const int* in_sizes, int n_in,
                              void* d_out, int out_size, void* d_ws, size_t ws_size,
                              hipStream_t stream) {
    (void)in_sizes; (void)n_in; (void)out_size; (void)ws_size;
    const float* x     = (const float*)d_in[0];
    /* d_in[1] edge_weight: unused by the reference computation */
    const float* W_enc = (const float*)d_in[2];
    const float* W_cls = (const float*)d_in[3];
    const int*   eidx  = (const int*)d_in[4];
    const int*   src   = eidx;          // row 0: node index
    const int*   he    = eidx + E_INC;  // row 1: hyperedge index

    float* out  = (float*)d_out;
    float* loss = out + (size_t)N_NODES * OUT_DIM;  // scalar after logits

    char* w = (char*)d_ws;
    float*    h         = (float*)w;    w += (size_t)N_NODES * HID * 4;
    float*    temp_edge = (float*)w;    w += (size_t)M_EDGES * HID * 4;
    float*    xe        = (float*)w;    w += (size_t)M_EDGES * F_DIM * 4;
    float*    x_out     = (float*)w;    w += (size_t)N_NODES * HEADS * F_DIM * 4;
    float*    cnt_he    = (float*)w;    w += (size_t)M_EDGES * 4;
    float*    cnt_node  = (float*)w;    w += (size_t)N_NODES * 4;
    unsigned* att_u     = (unsigned*)w; w += (size_t)M_EDGES * HEADS * 4;
    float*    att       = (float*)w;    w += (size_t)M_EDGES * HEADS * 4;
    v16bf*    bswz_enc  = (v16bf*)w;    w += (size_t)F_DIM * HID * 2;              // bf16
    v16bf*    bswz_cls  = (v16bf*)w;    w += (size_t)(HEADS * F_DIM) * OUT_DIM * 2;

    hipMemsetAsync(temp_edge, 0,    (size_t)M_EDGES * HID * 4, stream);
    hipMemsetAsync(xe,        0,    (size_t)M_EDGES * F_DIM * 4, stream);
    hipMemsetAsync(x_out,     0,    (size_t)N_NODES * HEADS * F_DIM * 4, stream);
    hipMemsetAsync(cnt_he,    0,    (size_t)M_EDGES * 4, stream);
    hipMemsetAsync(cnt_node,  0,    (size_t)N_NODES * 4, stream);
    hipMemsetAsync(att_u,     0xFF, (size_t)M_EDGES * HEADS * 4, stream);
    hipMemsetAsync(loss,      0,    4, stream);

    // 0. pre-swizzle weights into bf16 WMMA B-fragments
    {
        int t_enc = (F_DIM / 32) * (HID / 16);              // 128 tiles
        int t_cls = ((HEADS * F_DIM) / 32) * (OUT_DIM / 16);// 256 tiles
        k_pack_b<<<(t_enc + 7) / 8, 256, 0, stream>>>(W_enc, bswz_enc, F_DIM, HID);
        k_pack_b<<<(t_cls + 7) / 8, 256, 0, stream>>>(W_cls, bswz_cls, HEADS * F_DIM,
                                                      OUT_DIM);
    }
    // 1. encoder GEMM: h = x @ W_enc (WMMA bf16)
    {
        int tiles = (N_NODES / 16) * (HID / 16);
        k_gemm_bf16<<<(tiles + 7) / 8, 256, 0, stream>>>(x, bswz_enc, h, N_NODES,
                                                         F_DIM, HID);
    }
    // 2. per-head softmax over FS
    {
        int rows = N_NODES * HEADS;
        k_softmax<<<(rows + 7) / 8, 256, 0, stream>>>(h, rows);
    }
    // 3. incidence counts
    k_counts<<<(E_INC + 255) / 256, 256, 0, stream>>>(src, he, cnt_node, cnt_he, E_INC);
    // 4. temp_edge = segment_mean(h[src], he)
    k_scatter_add_rows<<<16384, 64, 0, stream>>>(h, src, he, temp_edge, E_INC);
    k_div_rows<<<M_EDGES, 256, 0, stream>>>(temp_edge, cnt_he, HID);
    // 5/6. attention: per-incidence head dots, scatter-min, decode (inf -> 0)
    k_att<<<E_INC, 256, 0, stream>>>(temp_edge, h, src, he, att_u, E_INC);
    k_att_decode<<<(M_EDGES * HEADS + 255) / 256, 256, 0, stream>>>(att_u, att,
                                                                    M_EDGES * HEADS);
    // 7. disentanglement factor loss
    k_factor_loss<<<(M_EDGES + 7) / 8, 256, 0, stream>>>(temp_edge, loss, M_EDGES);
    // 8. xe = segment_mean(x[src], he)
    k_scatter_add_rows<<<16384, 64, 0, stream>>>(x, src, he, xe, E_INC);
    k_div_rows<<<M_EDGES, 256, 0, stream>>>(xe, cnt_he, F_DIM);
    // 9/10. propagation: hyperedge -> node mean of att-scaled features
    k_prop<<<16384, 64, 0, stream>>>(xe, att, src, he, x_out, E_INC);
    k_div_rows<<<N_NODES, 256, 0, stream>>>(x_out, cnt_node, HEADS * F_DIM);
    // 11. classifier GEMM: logits = x_out @ W_cls (WMMA bf16)
    {
        int tiles = (N_NODES / 16) * (OUT_DIM / 16);
        k_gemm_bf16<<<(tiles + 7) / 8, 256, 0, stream>>>(x_out, bswz_cls, out, N_NODES,
                                                         HEADS * F_DIM, OUT_DIM);
    }
}
